// CumulativeShadeRegressor_76862734729481
// MI455X (gfx1250) — matrix operand order repaired and checked by reading, not compiled, not run on hardware
//
#include <hip/hip_runtime.h>
#include <math.h>

// Problem constants (from reference)
#define B_  256
#define L_  512
#define FD_ 64
#define G_  32
#define H1_ 512
#define H2_ 512
#define DH_ 256

typedef __attribute__((ext_vector_type(16))) _Float16     v16h;
typedef __attribute__((ext_vector_type(8)))  float        v8f;
typedef __attribute__((ext_vector_type(8)))  unsigned int v8u;

static __device__ __forceinline__ unsigned short f2h_bits(float x) {
  _Float16 h = (_Float16)x;
  return __builtin_bit_cast(unsigned short, h);
}
static __device__ __forceinline__ float h2f_bits(unsigned short u) {
  _Float16 h = __builtin_bit_cast(_Float16, u);
  return (float)h;
}
static __device__ __forceinline__ float gelu_f(float x) {
  return 0.5f * x * (1.0f + erff(x * 0.7071067811865475f));
}
static __device__ __forceinline__ float softplus_f(float x) {
  return (x > 20.0f) ? x : log1pf(expf(x));
}

// ---------------------------------------------------------------------------
// Workspace layout (bytes)
// ---------------------------------------------------------------------------
#define OFF_XL16   0
#define SZ_XL16    (B_ * L_ * FD_ * 2)        // Xl as f16            (16 MiB)
#define OFF_W1T    (OFF_XL16 + SZ_XL16)
#define SZ_W1T     (FD_ * H1_ * 2)            // W1^T f16  Wt1[n*64+k]
#define OFF_W2T    (OFF_W1T + SZ_W1T)
#define SZ_W2T     (H1_ * H2_ * 2)            // W2^T f16  Wt2[n*512+k]
#define OFF_POOLED (OFF_W2T + SZ_W2T)
#define SZ_POOLED  (B_ * H2_ * 4)             // f32 accum (atomic)
#define OFF_ABS    (OFF_POOLED + SZ_POOLED)
#define SZ_ABS     (B_ * L_ * 4)
#define OFF_ATT    (OFF_ABS + SZ_ABS)
#define SZ_ATT     (B_ * L_ * 4)

// ---------------------------------------------------------------------------
// Kernel 0: convert Xl -> f16, W1/W2 -> transposed f16, zero pooled accum
// ---------------------------------------------------------------------------
__global__ __launch_bounds__(256) void prep_kernel(
    const float* __restrict__ Xl, const float* __restrict__ W1,
    const float* __restrict__ W2,
    unsigned short* __restrict__ xl16, unsigned short* __restrict__ w1t,
    unsigned short* __restrict__ w2t, float* __restrict__ pooled) {
  const int i = blockIdx.x * blockDim.x + threadIdx.x;
  if (i < B_ * L_ * FD_) xl16[i] = f2h_bits(Xl[i]);
  if (i < FD_ * H1_) {                       // w1t[n*FD + k] = W1[k*H1 + n]
    const int n = i / FD_, k = i % FD_;
    w1t[i] = f2h_bits(W1[k * H1_ + n]);
  }
  if (i < H1_ * H2_) {                       // w2t[n*H1 + k] = W2[k*H2 + n]
    const int n = i / H1_, k = i % H1_;
    w2t[i] = f2h_bits(W2[k * H2_ + n]);
  }
  if (i < B_ * H2_) pooled[i] = 0.0f;
}

// ---------------------------------------------------------------------------
// Kernel 1: fused  h1 = gelu(Xl W1 + b1); h2 = gelu(h1 W2 + b2);
//           pooled += colsum(h2); absorb/atten = softplus(h2·wa/wt + b)
// One block = 64 consecutive rows of the flattened (B*L) x FD input
// (always within one batch since 64 | 512).
// 8 waves, each owning a 64-column slab (4 N-tiles) and ALL 64 rows
// (4 M-subtiles). Every B fragment loaded from L2 feeds FOUR WMMAs, and
// W1^T/W2^T are each read exactly once per workgroup (~1 GB total L2
// traffic for GEMM2, ~65 FLOP/byte vs L2). A fragments come from LDS.
// Accumulators: 4x4 tiles x 8 VGPR = 128 VGPRs/lane.
// ---------------------------------------------------------------------------
__global__ __launch_bounds__(256) void fused_main_kernel(
    const unsigned short* __restrict__ xl16,
    const unsigned short* __restrict__ w1t,
    const unsigned short* __restrict__ w2t,
    const float* __restrict__ b1, const float* __restrict__ b2,
    const float* __restrict__ wa, const float* __restrict__ ba,
    const float* __restrict__ wt, const float* __restrict__ bt,
    float* __restrict__ pooled, float* __restrict__ absorb,
    float* __restrict__ atten) {
  __shared__ unsigned short lds_h[64 * H1_];  // 64 KiB: h1 then h2 (f16)

  const int tid  = threadIdx.x;
  const int lane = tid & 31;
  const int nw   = tid >> 5;     // wave id = N-column slab (0..7)
  const int lo   = lane & 15;    // row (A) / col (B,C) selector
  const int hi   = lane >> 4;    // K-half selector

  const int rowBase = blockIdx.x * 64;        // flattened row base
  const int b       = rowBase >> 9;           // batch index
  const int l0      = rowBase & (L_ - 1);     // L offset within batch

  // ------------------ GEMM1: 64x64 tile of h1 = gelu(Xl W1 + b1) -----------
  v8f acc[4][4] = {};
#pragma unroll
  for (int ks = 0; ks < 2; ++ks) {            // K = 64 -> 2 steps of 32
    const int kStart = ks * 32;
    v16h afr[4];
#pragma unroll
    for (int ms = 0; ms < 4; ++ms) {
      const int aRow = rowBase + ms * 16 + lo;
      v8u au;
#pragma unroll
      for (int v = 0; v < 8; ++v) {
        const int kb = (v < 4) ? (2 * v + 8 * hi) : (16 + 2 * (v - 4) + 8 * hi);
        au[v] = *(const unsigned int*)(xl16 + aRow * FD_ + kStart + kb);
      }
      afr[ms] = __builtin_bit_cast(v16h, au);
    }
#pragma unroll
    for (int t = 0; t < 4; ++t) {
      const int n = nw * 64 + t * 16 + lo;
      v8u bu;
#pragma unroll
      for (int v = 0; v < 8; ++v)
        bu[v] = *(const unsigned int*)(w1t + n * FD_ + kStart + ((v < 4) ? (2 * v + 8 * hi) : (16 + 2 * (v - 4) + 8 * hi)));
      const v16h bfr = __builtin_bit_cast(v16h, bu);
#pragma unroll
      for (int ms = 0; ms < 4; ++ms)
        acc[ms][t] = __builtin_amdgcn_wmma_f32_16x16x32_f16(
            false, afr[ms], false, bfr, (short)0, acc[ms][t], false, false);
    }
  }
  // epilogue: bias + gelu, store f16 into LDS (C layout: M = r + 8*hi, N = lo)
#pragma unroll
  for (int ms = 0; ms < 4; ++ms) {
#pragma unroll
    for (int t = 0; t < 4; ++t) {
      const int n    = nw * 64 + t * 16 + lo;
      const float bb = b1[n];
#pragma unroll
      for (int r = 0; r < 8; ++r) {
        const int m = ms * 16 + r + 8 * hi;
        lds_h[m * H1_ + n] = f2h_bits(gelu_f(acc[ms][t][r] + bb));
      }
    }
  }
  __syncthreads();

  // ------------------ GEMM2: 64x64 tile of h2 = gelu(h1 W2 + b2) -----------
#pragma unroll
  for (int ms = 0; ms < 4; ++ms)
#pragma unroll
    for (int t = 0; t < 4; ++t)
#pragma unroll
      for (int r = 0; r < 8; ++r) acc[ms][t][r] = 0.0f;

  for (int ks = 0; ks < 16; ++ks) {           // K = 512 -> 16 steps
    const int kStart = ks * 32;
    v16h afr[4];
#pragma unroll
    for (int ms = 0; ms < 4; ++ms) {
      const int aRow = ms * 16 + lo;
      v8u au;
#pragma unroll
      for (int v = 0; v < 8; ++v) {
        const int kb = (v < 4) ? (2 * v + 8 * hi) : (16 + 2 * (v - 4) + 8 * hi);
        au[v] = *(const unsigned int*)&lds_h[aRow * H1_ + kStart + kb];
      }
      afr[ms] = __builtin_bit_cast(v16h, au);
    }
#pragma unroll
    for (int t = 0; t < 4; ++t) {
      const int n = nw * 64 + t * 16 + lo;
      v8u bu;
#pragma unroll
      for (int v = 0; v < 8; ++v)
        bu[v] = *(const unsigned int*)(w2t + n * H1_ + kStart + ((v < 4) ? (2 * v + 8 * hi) : (16 + 2 * (v - 4) + 8 * hi)));
      const v16h bfr = __builtin_bit_cast(v16h, bu);
#pragma unroll
      for (int ms = 0; ms < 4; ++ms)
        acc[ms][t] = __builtin_amdgcn_wmma_f32_16x16x32_f16(
            false, afr[ms], false, bfr, (short)0, acc[ms][t], false, false);
    }
  }
  // epilogue: bias + gelu -> h2; accumulate pooled column sums
#pragma unroll
  for (int ms = 0; ms < 4; ++ms) {
#pragma unroll
    for (int t = 0; t < 4; ++t) {
      const int n    = nw * 64 + t * 16 + lo;
      const float bb = b2[n];
      float colsum = 0.0f;
#pragma unroll
      for (int r = 0; r < 8; ++r) {
        const float g = gelu_f(acc[ms][t][r] + bb);
        acc[ms][t][r] = g;
        colsum += g;
      }
      atomicAdd(&pooled[b * H2_ + n], colsum);
    }
  }
  __syncthreads();                            // all waves done reading h1
#pragma unroll
  for (int ms = 0; ms < 4; ++ms) {
#pragma unroll
    for (int t = 0; t < 4; ++t) {
      const int n = nw * 64 + t * 16 + lo;
#pragma unroll
      for (int r = 0; r < 8; ++r) {
        const int m = ms * 16 + r + 8 * hi;
        lds_h[m * H2_ + n] = f2h_bits(acc[ms][t][r]);   // overwrite with h2
      }
    }
  }
  __syncthreads();

  // ------------------ absorb / atten: per-row 512-length dots --------------
  const int row = tid >> 2;                   // 0..63
  const int seg = tid & 3;                    // 4 threads per row
  float da = 0.0f, dt = 0.0f;
  for (int c = seg * 128; c < seg * 128 + 128; ++c) {
    const float hv = h2f_bits(lds_h[row * H2_ + c]);
    da += hv * wa[c];
    dt += hv * wt[c];
  }
#pragma unroll
  for (int off = 2; off >= 1; off >>= 1) {
    da += __shfl_down(da, off);
    dt += __shfl_down(dt, off);
  }
  if (seg == 0) {
    absorb[b * L_ + l0 + row] = softplus_f(da + ba[0]);
    atten[b * L_ + l0 + row]  = softplus_f(dt + bt[0]);
  }
}

// ---------------------------------------------------------------------------
// Kernel 2: per-batch reversed exclusive cumsum + captured; MLP head; output
// ---------------------------------------------------------------------------
__global__ __launch_bounds__(256) void finalize_kernel(
    const float* __restrict__ Xg, const float* __restrict__ pooled,
    const float* __restrict__ absorb, const float* __restrict__ atten,
    const float* __restrict__ Wd1, const float* __restrict__ bd1,
    const float* __restrict__ Wd2, const float* __restrict__ bd2,
    float* __restrict__ out) {
  __shared__ float s_att[L_];
  __shared__ float s_abs[L_];
  __shared__ float s_red[256];
  __shared__ float s_z[G_ + H2_];
  __shared__ float s_cap;

  const int b   = blockIdx.x;
  const int tid = threadIdx.x;

  for (int i = tid; i < L_; i += 256) {       // time-reversed load
    s_att[i] = atten[b * L_ + (L_ - 1 - i)];
    s_abs[i] = absorb[b * L_ + (L_ - 1 - i)];
  }
  __syncthreads();
  if (tid == 0) {                             // exclusive scan (serial, cheap)
    float run = 0.0f;
    for (int i = 0; i < L_; ++i) {
      const float a = s_att[i];
      s_att[i] = run;
      run += a;
    }
  }
  __syncthreads();
  float part = 0.0f;
  for (int i = tid; i < L_; i += 256) part += expf(-s_att[i]) * s_abs[i];
  s_red[tid] = part;
  __syncthreads();
  for (int s = 128; s > 0; s >>= 1) {
    if (tid < s) s_red[tid] += s_red[tid + s];
    __syncthreads();
  }
  if (tid == 0) s_cap = s_red[0];             // captured

  // z = concat(Xg[b], mean_L(h2))
  if (tid < G_) s_z[tid] = Xg[b * G_ + tid];
  for (int h = tid; h < H2_; h += 256)
    s_z[G_ + h] = pooled[b * H2_ + h] * (1.0f / (float)L_);
  __syncthreads();

  // d1[tid] = gelu(z . Wd1[:,tid] + bd1[tid]);  partial = d1*Wd2
  float accv = bd1[tid];
  for (int k = 0; k < G_ + H2_; ++k) accv += s_z[k] * Wd1[k * DH_ + tid];
  const float g = gelu_f(accv) * Wd2[tid];
  __syncthreads();
  s_red[tid] = g;
  __syncthreads();
  for (int s = 128; s > 0; s >>= 1) {
    if (tid < s) s_red[tid] += s_red[tid + s];
    __syncthreads();
  }
  if (tid == 0) out[b] = s_cap + s_red[0] + bd2[0];
}

// ---------------------------------------------------------------------------
extern "C" void kernel_launch(void* const* d_in, const int* in_sizes, int n_in,
                              void* d_out, int out_size, void* d_ws,
                              size_t ws_size, hipStream_t stream) {
  (void)in_sizes; (void)n_in; (void)out_size; (void)ws_size;
  const float* Xg  = (const float*)d_in[0];
  const float* Xl  = (const float*)d_in[1];
  const float* W1  = (const float*)d_in[2];
  const float* b1  = (const float*)d_in[3];
  const float* W2  = (const float*)d_in[4];
  const float* b2  = (const float*)d_in[5];
  const float* wa  = (const float*)d_in[6];
  const float* ba  = (const float*)d_in[7];
  const float* wt  = (const float*)d_in[8];
  const float* bt  = (const float*)d_in[9];
  const float* Wd1 = (const float*)d_in[10];
  const float* bd1 = (const float*)d_in[11];
  const float* Wd2 = (const float*)d_in[12];
  const float* bd2 = (const float*)d_in[13];
  float* out = (float*)d_out;

  char* ws = (char*)d_ws;
  unsigned short* xl16 = (unsigned short*)(ws + OFF_XL16);
  unsigned short* w1t  = (unsigned short*)(ws + OFF_W1T);
  unsigned short* w2t  = (unsigned short*)(ws + OFF_W2T);
  float* pooled        = (float*)(ws + OFF_POOLED);
  float* absorb        = (float*)(ws + OFF_ABS);
  float* atten         = (float*)(ws + OFF_ATT);

  prep_kernel<<<(B_ * L_ * FD_ + 255) / 256, 256, 0, stream>>>(
      Xl, W1, W2, xl16, w1t, w2t, pooled);
  fused_main_kernel<<<(B_ * L_) / 64, 256, 0, stream>>>(
      xl16, w1t, w2t, b1, b2, wa, ba, wt, bt, pooled, absorb, atten);
  finalize_kernel<<<B_, 256, 0, stream>>>(
      Xg, pooled, absorb, atten, Wd1, bd1, Wd2, bd2, out);
}
